// CAML_49735721288067
// MI455X (gfx1250) — matrix operand order repaired and checked by reading, not compiled, not run on hardware
//
#include <hip/hip_runtime.h>
#include <hip/hip_bf16.h>

typedef __attribute__((ext_vector_type(16))) _Float16 v16h;
typedef __attribute__((ext_vector_type(8)))  float    v8f;

#define Bq   8
#define Wlen 1000
#define Vv   50000
#define Ee   100
#define Cc   50
#define Kk   3
#define Ll   18000
#define CP   64    /* C padded to 64 (WMMA K for scores GEMM)         */
#define WP   1024  /* W padded to 1024 (WMMA K stepping for m GEMM)   */

union V16H {
    v16h     v;
    _Float16 h[16];
    unsigned u[8];
    uint4    q[2];
};

// workspace layout (bytes):
//   Hrow : [B][WP][CP] f16  at 0            (w-major, c contiguous)  1 MB + slack
//   Hcol : [B][CP][WP] f16  at HCOL_OFF     (c-major, w contiguous)  1 MB
#define HROW_ELEMS   ((size_t)Bq * WP * CP)
#define HCOL_OFF     ((HROW_ELEMS + 16 * CP) * sizeof(_Float16))  /* + prefetch slack */

// ---------------------------------------------------------------------------
// Kernel 1: embedding gather + conv1d(K=3, same pad) + bias + tanh.
// Writes BOTH layouts of H (f16), with w in [1000,1024) zero-filled so the
// attention kernel's WMMA operand loads are completely unconditional.
// ---------------------------------------------------------------------------
__global__ __launch_bounds__(64) void caml_conv_tanh_kernel(
    const int*   __restrict__ x,        // [B][W]
    const float* __restrict__ W_embed,  // [V][E]
    const float* __restrict__ conv_w,   // [C][E][K]
    const float* __restrict__ conv_b,   // [C]
    _Float16*    __restrict__ Hrow,     // [B][WP][CP]
    _Float16*    __restrict__ Hcol)     // [B][CP][WP]
{
    __shared__ float emb[Kk][Ee];
    const int b   = blockIdx.x >> 10;
    const int w   = blockIdx.x & (WP - 1);
    const int tid = threadIdx.x;

    _Float16 hv = (_Float16)0.f;
    if (w < Wlen) {
        // cooperative load of the 3 neighboring embedding rows (zero outside)
        for (int i = tid; i < Kk * Ee; i += 64) {
            const int k = i / Ee, e = i % Ee;
            const int wsrc = w + k - (Kk / 2);
            float v = 0.f;
            if (wsrc >= 0 && wsrc < Wlen) {
                const int idx = x[b * Wlen + wsrc];
                v = W_embed[(size_t)idx * Ee + e];
            }
            emb[k][e] = v;
        }
        __syncthreads();

        const int c = tid;
        if (c < Cc) {
            float acc = conv_b[c];
            const float* wc = conv_w + (size_t)c * Ee * Kk;  // [e][k]
            #pragma unroll 4
            for (int e = 0; e < Ee; ++e) {
                acc += emb[0][e] * wc[e * Kk + 0];
                acc += emb[1][e] * wc[e * Kk + 1];
                acc += emb[2][e] * wc[e * Kk + 2];
            }
            hv = (_Float16)tanhf(acc);
        }
    }
    // all 64 threads write both layouts (zeros for c>=50 or w>=1000 pads)
    Hrow[((size_t)b * WP + w) * CP + tid] = hv;
    Hcol[((size_t)b * CP + tid) * WP + w] = hv;
}

// ---------------------------------------------------------------------------
// Kernel 2: fused label-wise attention for a (batch, 16-label) tile.
// One wave32 per block. scores -> LDS -> softmax -> attn@H -> out_w dot.
// ---------------------------------------------------------------------------
__global__ __launch_bounds__(32) void caml_attn_kernel(
    const _Float16* __restrict__ Hrow,  // [B][WP][CP]
    const _Float16* __restrict__ Hcol,  // [B][CP][WP]
    const float*    __restrict__ u_w,   // [L][C]
    const float*    __restrict__ out_w, // [L][C]
    const float*    __restrict__ out_b, // [L]
    float*          __restrict__ out)   // [B][L]
{
    __shared__ float sc[16 * WP];  // 64 KB; reused: u_w stage -> scores -> m
    const int lane = threadIdx.x;
    const int blk  = blockIdx.x;
    const int b    = blk / (Ll / 16);
    const int l0   = (blk % (Ll / 16)) * 16;
    const int nmod = lane & 15;
    const int nhi  = lane >> 4;

    // --- stage u_w tile in LDS (divergence-free: 1024 and 800 are /32) -----
    for (int i = lane; i < 16 * 64; i += 32) sc[i] = 0.f;        // 32 iters
    __syncthreads();
    for (int i = lane; i < 16 * Cc; i += 32) {                   // 25 iters
        const int row = i / Cc, c = i % Cc;
        sc[row * 64 + c] = u_w[(size_t)(l0 + row) * Cc + c];
    }
    __syncthreads();

    // --- build the two A operands from LDS ---------------------------------
    // A k order used here (B the same): lane -> M=lane%16; element e in group g=e/8:
//   k = (lane/16)*16 + g*8 + (e - g*8)
    V16H a0, a1;
    {
        const float* ur = sc + nmod * 64;
        #pragma unroll
        for (int g = 0; g < 2; ++g)
            #pragma unroll
            for (int j = 0; j < 8; ++j) {
                const int k = nhi * 16 + g * 8 + j;
                a0.h[g * 8 + j] = (_Float16)ur[k];
                a1.h[g * 8 + j] = (_Float16)ur[k + 32];
            }
    }
    __syncthreads();

    const _Float16* Hb  = Hrow + (size_t)b * WP * CP;
    const _Float16* Htb = Hcol + (size_t)b * CP * WP;

    // --- Phase 1: scores[16 l][1024 w] via WMMA, 64 w-tiles of 16 ----------
    for (int wt = 0; wt < 64; ++wt) {
        const int w = wt * 16 + nmod;               // lane -> N = w column
        const uint4* hq = (const uint4*)(Hb + (size_t)w * CP);
        __builtin_prefetch(Hb + (size_t)(w + 16) * CP, 0, 0);
        V16H b0, b1;                                 // K = c
        b0.q[0] = hq[nhi * 2 + 0];
        b0.q[1] = hq[nhi * 2 + 1];
        b1.q[0] = hq[nhi * 2 + 4];
        b1.q[1] = hq[nhi * 2 + 5];
        v8f acc = {};
        acc = __builtin_amdgcn_wmma_f32_16x16x32_f16(false, a0.v, false, b0.v,
                                                     (short)0, acc, false, false);
        acc = __builtin_amdgcn_wmma_f32_16x16x32_f16(false, a1.v, false, b1.v,
                                                     (short)0, acc, false, false);
        // C/D layout: lane -> N=w, VGPR r -> M=(lane/16)*8+r. Pad w -> -inf
        // (branch-free cndmask; pad H rows are zero so WMMA result is junk-free)
        const float pad = -1e30f;
        const bool wok = (w < Wlen);
        #pragma unroll
        for (int r = 0; r < 8; ++r) {
            const int M = nhi * 8 + r;
            sc[M * WP + w] = wok ? acc[r] : pad;
        }
    }
    __syncthreads();

    // --- Phase 2: softmax over w per label row (f32, float4 LDS traffic) ---
    // lane handles row = lane%16, half = lane/16 (512 entries each).
    float invDenom;
    {
        float4* rowq = (float4*)(sc + nmod * WP + nhi * 512);
        float mx = -1e30f;
        for (int i = 0; i < 128; ++i) {
            const float4 f = rowq[i];
            mx = fmaxf(mx, fmaxf(fmaxf(f.x, f.y), fmaxf(f.z, f.w)));
        }
        mx = fmaxf(mx, __shfl_xor(mx, 16, 32));
        float s = 0.f;
        for (int i = 0; i < 128; ++i) {
            float4 f = rowq[i];
            f.x = __expf(f.x - mx);
            f.y = __expf(f.y - mx);
            f.z = __expf(f.z - mx);
            f.w = __expf(f.w - mx);
            rowq[i] = f;  // keep numerators in LDS
            s += f.x + f.y + f.z + f.w;
        }
        s += __shfl_xor(s, 16, 32);
        invDenom = 1.f / s;  // lane-local; lane's A-row == its softmax row
    }
    __syncthreads();

    // --- Phase 3: m[16 l][64 c] = attn @ H, K = w in 32 steps of 32 --------
    v8f macc0 = {}, macc1 = {}, macc2 = {}, macc3 = {};
    for (int ws = 0; ws < 32; ++ws) {
        // A = attn tile 16(l) x 32(w): f32 numerators from LDS * invDenom
        V16H at;
        {
            const float* row = sc + nmod * WP + ws * 32;
            #pragma unroll
            for (int g = 0; g < 2; ++g) {
                const float4 f0 = *(const float4*)(row + nhi * 16 + g * 8);
                const float4 f1 = *(const float4*)(row + nhi * 16 + g * 8 + 4);
                at.h[g * 8 + 0] = (_Float16)(f0.x * invDenom);
                at.h[g * 8 + 1] = (_Float16)(f0.y * invDenom);
                at.h[g * 8 + 2] = (_Float16)(f0.z * invDenom);
                at.h[g * 8 + 3] = (_Float16)(f0.w * invDenom);
                at.h[g * 8 + 4] = (_Float16)(f1.x * invDenom);
                at.h[g * 8 + 5] = (_Float16)(f1.y * invDenom);
                at.h[g * 8 + 6] = (_Float16)(f1.z * invDenom);
                at.h[g * 8 + 7] = (_Float16)(f1.w * invDenom);
            }
        }
        #pragma unroll
        for (int ct = 0; ct < 4; ++ct) {
            // B = H tile 32(w) x 16(c): lane -> c column; K=w contiguous in Hcol
            const int ccol = ct * 16 + nmod;
            const uint4* hq =
                (const uint4*)(Htb + (size_t)ccol * WP + ws * 32 + nhi * 16);
            V16H bt;
            bt.q[0] = hq[0];
            bt.q[1] = hq[1];
            v8f acc = (ct == 0) ? macc0 : (ct == 1) ? macc1
                    : (ct == 2) ? macc2 : macc3;
            acc = __builtin_amdgcn_wmma_f32_16x16x32_f16(false, at.v, false, bt.v,
                                                         (short)0, acc, false, false);
            if      (ct == 0) macc0 = acc;
            else if (ct == 1) macc1 = acc;
            else if (ct == 2) macc2 = acc;
            else              macc3 = acc;
        }
    }
    __syncthreads();

    // --- Phase 4: spill m to LDS (reuse score buffer), final dot + sigmoid -
    #pragma unroll
    for (int r = 0; r < 8; ++r) {
        const int M = nhi * 8 + r;
        sc[M * 64 +  0 + nmod] = macc0[r];
        sc[M * 64 + 16 + nmod] = macc1[r];
        sc[M * 64 + 32 + nmod] = macc2[r];
        sc[M * 64 + 48 + nmod] = macc3[r];
    }
    __syncthreads();
    if (lane < 16) {
        const int label = l0 + lane;
        const float* ow = out_w + (size_t)label * Cc;
        float dot = out_b[label];
        for (int c = 0; c < Cc; ++c) dot += ow[c] * sc[lane * 64 + c];
        out[(size_t)b * Ll + label] = 1.f / (1.f + __expf(-dot));
    }
}

// ---------------------------------------------------------------------------
extern "C" void kernel_launch(void* const* d_in, const int* in_sizes, int n_in,
                              void* d_out, int out_size, void* d_ws, size_t ws_size,
                              hipStream_t stream) {
    const int*   x       = (const int*)d_in[0];
    const float* W_embed = (const float*)d_in[1];
    const float* conv_w  = (const float*)d_in[2];
    const float* conv_b  = (const float*)d_in[3];
    const float* u_w     = (const float*)d_in[4];
    const float* out_w   = (const float*)d_in[5];
    const float* out_b   = (const float*)d_in[6];
    float*       out     = (float*)d_out;

    _Float16* Hrow = (_Float16*)d_ws;
    _Float16* Hcol = (_Float16*)((char*)d_ws + HCOL_OFF);

    caml_conv_tanh_kernel<<<Bq * WP, 64, 0, stream>>>(
        x, W_embed, conv_w, conv_b, Hrow, Hcol);

    caml_attn_kernel<<<Bq * (Ll / 16), 32, 0, stream>>>(
        Hrow, Hcol, u_w, out_w, out_b, out);
}
